// DFN1_42442866819723
// MI455X (gfx1250) — matrix operand order repaired and checked by reading, not compiled, not run on hardware
//
#include <hip/hip_runtime.h>

typedef __attribute__((ext_vector_type(16))) _Float16 v16h;
typedef __attribute__((ext_vector_type(8)))  float    v8f;

// ---- WMMA fragment index maps (CDNA5 ISA 7.12.2, wave32) ------------------
// A (16x32 f16): lane(0-15)=row M; VGPR0-3 hold K0-7 (upper lanes K8-15),
//                VGPR4-7 hold K16-23 (upper lanes K24-31).
__device__ __forceinline__ int kmapA(int j, int half) {
  return (j < 8) ? (j + 8 * half) : (j + 8 + 8 * half);
}
// B (32x16 f16): same K order as A: lanes 0-15 hold K0-7,16-23; lanes 16-31 K8-15,24-31.
__device__ __forceinline__ int kmapB(int j, int half) {
  return (j < 8) ? (j + 8 * half) : (j + 8 + 8 * half);
}

// ---------------------------------------------------------------------------
// Pack conv weights (f32, OIHW or transposed IOHW+flip) into f16 B-fragments.
// Layout: v16h element ((nt*nK + kt)*32 + lane) -> contiguous 32B per lane.
// mode 0: normal conv, w[Cout][Cin][Kk][Kk]
// mode 1: ConvTranspose2d, w[Cin][Cout][Kk][Kk], kernel spatially flipped.
// Runs once per launch (weights are timestep-invariant) -> generic is fine.
// ---------------------------------------------------------------------------
__global__ __launch_bounds__(32)
void pack_w_kernel(const float* __restrict__ w, _Float16* __restrict__ bp,
                   int Cout, int Cin, int Kk, int mode) {
  const int KkK   = Kk * Kk;
  const int Kreal = Cin * KkK;
  const int nK    = (Kreal + 31) >> 5;
  const int lane  = threadIdx.x;
  const int kt    = blockIdx.x;
  const int nt    = blockIdx.y;
  const int half  = lane >> 4;
  const int n     = nt * 16 + (lane & 15);
  v16h bv;
#pragma unroll
  for (int j = 0; j < 16; ++j) {
    int k = kt * 32 + kmapB(j, half);
    float v = 0.f;
    if (n < Cout && k < Kreal) {
      int ci = k / KkK;
      int r  = k - ci * KkK;
      int ky = r / Kk;
      int kx = r - ky * Kk;
      if (mode == 0)
        v = w[((n * Cin + ci) * Kk + ky) * Kk + kx];
      else
        v = w[((ci * Cout + n) * Kk + (Kk - 1 - ky)) * Kk + (Kk - 1 - kx)];
    }
    bv[j] = (_Float16)v;
  }
  ((v16h*)bp)[((size_t)nt * nK + kt) * 32 + lane] = bv;
}

// ---------------------------------------------------------------------------
// im2col pack: write the activation GEMM operand in WMMA A-fragment order.
// One thread builds one v16h (16 gathered f16). All shape params are
// compile-time -> every div/mod is by a constant (magic-multiply, no v_rcp).
// mode 0: conv(x, stride=sd, pad=pq)
// mode 1: dilated-input gather (ConvTranspose2d, pre-flipped kernel in B).
// grid = (nK, M/16), block = 32.
// ---------------------------------------------------------------------------
template <int Cin, int IH, int OH, int Kk, int sd, int pq, int mode>
__global__ __launch_bounds__(32)
void pack_a_kernel(const float* __restrict__ x, _Float16* __restrict__ ap) {
  constexpr int IW    = IH;
  constexpr int OW    = OH;
  constexpr int OHW   = OH * OW;
  constexpr int KkK   = Kk * Kk;
  constexpr int Kreal = Cin * KkK;
  constexpr int nK    = (Kreal + 31) >> 5;

  const int lane = threadIdx.x;
  const int kt   = blockIdx.x;
  const int mt   = blockIdx.y;
  const int half = lane >> 4;
  const int m    = mt * 16 + (lane & 15);

  const int b  = m / OHW;
  const int r0 = m - b * OHW;
  const int oy = r0 / OW;
  const int ox = r0 - oy * OW;
  const float* xb = x + (size_t)b * Cin * IH * IW;

  v16h a;
#pragma unroll
  for (int j = 0; j < 16; ++j) {
    int k = kt * 32 + kmapA(j, half);
    float v = 0.f;
    if (k < Kreal) {
      int ci = k / KkK;
      int r  = k - ci * KkK;
      int ky = r / Kk;
      int kx = r - ky * Kk;
      if (mode == 0) {
        int iy = oy * sd - pq + ky;
        int ix = ox * sd - pq + kx;
        if ((unsigned)iy < (unsigned)IH && (unsigned)ix < (unsigned)IW)
          v = xb[(ci * IH + iy) * IW + ix];
      } else {
        int vy = oy - pq + ky;
        int vx = ox - pq + kx;
        if (vy >= 0 && vx >= 0 && (vy % sd) == 0 && (vx % sd) == 0) {
          int iy = vy / sd, ix = vx / sd;
          if (iy < IH && ix < IW) v = xb[(ci * IH + iy) * IW + ix];
        }
      }
    }
    a[j] = (_Float16)v;
  }
  ((v16h*)ap)[((size_t)mt * nK + kt) * 32 + lane] = a;
}

// ---------------------------------------------------------------------------
// Pure WMMA GEMM on pre-packed fragments: one wave32 owns a 32(M)x16(N) tile
// (2 accumulators sharing one B fragment -> 2 independent v_wmma per K-step).
// Fused epilogue: bias -> LeakyReLU(0.01) -> BatchNorm(eval) [-> +addp]
// scattered back to NCHW. grid = (M/32, ceil(Cout/16)), block = 32.
// ---------------------------------------------------------------------------
template <int Cout, int OH>
__global__ __launch_bounds__(32)
void gemm_wmma_kernel(const _Float16* __restrict__ ap, const _Float16* __restrict__ bp,
                      const float* __restrict__ bias,
                      const float* __restrict__ bng, const float* __restrict__ bnb,
                      const float* __restrict__ bnm, const float* __restrict__ bnv,
                      const float* __restrict__ addp,
                      float* __restrict__ out, int nK) {
  constexpr int OW  = OH;
  constexpr int OHW = OH * OW;
  const int lane = threadIdx.x;
  const int half = lane >> 4;
  const int l15  = lane & 15;
  const int bx   = blockIdx.x;
  const int nt   = blockIdx.y;

  const v16h* A0 = (const v16h*)ap + ((size_t)(2 * bx) * nK) * 32 + lane;
  const v16h* A1 = A0 + (size_t)nK * 32;
  const v16h* Bv = (const v16h*)bp + ((size_t)nt * nK) * 32 + lane;

  v8f acc0 = {}, acc1 = {};
  for (int kt = 0; kt < nK; ++kt) {
    v16h a0 = A0[(size_t)kt * 32];
    v16h a1 = A1[(size_t)kt * 32];
    v16h bb = Bv[(size_t)kt * 32];
    __builtin_prefetch((const void*)(A0 + (size_t)(kt + 1) * 32), 0, 1);
    __builtin_prefetch((const void*)(A1 + (size_t)(kt + 1) * 32), 0, 1);
    acc0 = __builtin_amdgcn_wmma_f32_16x16x32_f16(false, a0, false, bb,
                                                  (short)0, acc0, false, false);
    acc1 = __builtin_amdgcn_wmma_f32_16x16x32_f16(false, a1, false, bb,
                                                  (short)0, acc1, false, false);
  }

  const int n = nt * 16 + l15;
  if (n < Cout) {
    const float bi  = bias[n];
    const float inv = bng[n] * rsqrtf(bnv[n] + 1e-5f);
    const float mu  = bnm[n];
    const float be  = bnb[n];
#pragma unroll
    for (int sub = 0; sub < 2; ++sub) {
      v8f acc = sub ? acc1 : acc0;
#pragma unroll
      for (int r = 0; r < 8; ++r) {
        int mr  = bx * 32 + sub * 16 + r + 8 * half;
        int b2  = mr / OHW;             // constant divisor
        int rr  = mr - b2 * OHW;
        int oy2 = rr / OW;              // constant divisor
        int ox2 = rr - oy2 * OW;
        float y = acc[r] + bi;
        y = (y >= 0.f) ? y : 0.01f * y;
        y = (y - mu) * inv + be;
        size_t oidx = (((size_t)b2 * Cout + n) * OH + oy2) * OW + ox2;
        if (addp) y += addp[oidx];
        out[oidx] = y;
      }
    }
  }
}

// ---------------------------------------------------------------------------
// Softmax over 81 decode channels (through the (o*57)//64 nearest-resample
// index map) fused with the 9x9 dynamic-filter application on `frame`.
// ---------------------------------------------------------------------------
__global__ __launch_bounds__(256)
void dfn_filter_kernel(const float* __restrict__ d3, const float* __restrict__ frame,
                       float* __restrict__ outp, int Bn) {
  int t = blockIdx.x * blockDim.x + threadIdx.x;
  int total = Bn * 4096;
  if (t >= total) return;
  int b = t >> 12;
  int p = t & 4095;
  int y = p >> 6;
  int x = p & 63;
  int iy = (y * 57) >> 6;
  int ix = (x * 57) >> 6;
  const float* base = d3 + (size_t)b * 81 * 3249 + (size_t)iy * 57 + ix;
  const float* fb   = frame + (size_t)b * 4096;

  float mx = -3.4e38f;
  for (int c = 0; c < 81; ++c) mx = fmaxf(mx, base[(size_t)c * 3249]);
  float s = 0.f, acc = 0.f;
  for (int c = 0; c < 81; ++c) {
    float e = __expf(base[(size_t)c * 3249] - mx);
    s += e;
    int ky = c / 9;
    int kx = c - ky * 9;
    int sy = y + ky - 4;
    int sx = x + kx - 4;
    float fv = ((unsigned)sy < 64u && (unsigned)sx < 64u) ? fb[(sy << 6) + sx] : 0.f;
    acc += fv * e;
  }
  outp[t] = acc / s;
}

// ---- frame gather: input_frames is (B,1,64,64,T) -> frame i is strided ----
__global__ __launch_bounds__(256)
void gather_frame_kernel(const float* __restrict__ in, float* __restrict__ f,
                         int i, int T, int n) {
  int t = blockIdx.x * blockDim.x + threadIdx.x;
  if (t < n) f[t] = in[(size_t)t * T + i];
}

__global__ __launch_bounds__(256)
void zero_kernel(float* __restrict__ p, int n) {
  int t = blockIdx.x * blockDim.x + threadIdx.x;
  if (t < n) p[t] = 0.f;
}

// ---------------------------------------------------------------------------
extern "C" void kernel_launch(void* const* d_in, const int* in_sizes, int n_in,
                              void* d_out, int out_size, void* d_ws, size_t ws_size,
                              hipStream_t stream) {
  (void)in_sizes; (void)n_in; (void)out_size; (void)ws_size;
  const float* frames = (const float*)d_in[0];

  // params flattened in dict order: per layer {w, b, bn:(g,b,m,v)} x 8 layers
  const float *W[8], *Bi[8], *Gg[8], *Gb[8], *Gm[8], *Gv[8];
  for (int l = 0; l < 8; ++l) {
    int k = 1 + l * 6;
    W[l]  = (const float*)d_in[k + 0];
    Bi[l] = (const float*)d_in[k + 1];
    Gg[l] = (const float*)d_in[k + 2];
    Gb[l] = (const float*)d_in[k + 3];
    Gm[l] = (const float*)d_in[k + 4];
    Gv[l] = (const float*)d_in[k + 5];
  }
  // d_in[49]/d_in[50] are device scalars (10/10 per setup_inputs).
  const int NIN = 10, NOUT = 10, T = 10;
  const int Bn = 64;

  // layer order: en1 en2 en3 hc1 hc2 de1 de2 de3 (for weight packing)
  const int CinL [8] = {1, 32, 64, 128, 128, 128, 64, 32};
  const int CoutL[8] = {32, 64, 128, 128, 128, 64, 32, 81};
  const int KkL  [8] = {9, 9, 3, 3, 3, 3, 3, 3};
  const int modeL[8] = {0, 0, 0, 0, 0, 1, 1, 1};

  // ---- workspace carve ----
  char* p = (char*)d_ws;
  auto carve = [&](size_t bytes) -> char* {
    char* r = p;
    p += (bytes + 255) & ~(size_t)255;
    return r;
  };
  float* Fbuf = (float*)carve((size_t)Bn * 4096 * 4);
  float* E1   = (float*)carve((size_t)Bn * 32 * 32 * 32 * 4);
  float* E2   = (float*)carve((size_t)Bn * 64 * 16 * 16 * 4);
  float* E3   = (float*)carve((size_t)Bn * 128 * 8 * 8 * 4);
  float* H1   = (float*)carve((size_t)Bn * 128 * 8 * 8 * 4);
  float* Hid  = (float*)carve((size_t)Bn * 128 * 8 * 8 * 4);
  float* D1   = (float*)carve((size_t)Bn * 64 * 15 * 15 * 4);
  float* D2   = (float*)carve((size_t)Bn * 32 * 29 * 29 * 4);
  float* D3   = (float*)carve((size_t)Bn * 81 * 57 * 57 * 4);
  _Float16* PK[8];
  int nKL[8], nTL[8];
  size_t abytes = 0;
  for (int l = 0; l < 8; ++l) {
    int Kreal = CinL[l] * KkL[l] * KkL[l];
    nKL[l] = (Kreal + 31) >> 5;
    nTL[l] = (CoutL[l] + 15) >> 4;
    PK[l]  = (_Float16*)carve((size_t)nTL[l] * nKL[l] * 32 * 16 * sizeof(_Float16));
    size_t need = (size_t)Bn * ((l == 5) ? 15 * 15 : (l == 6) ? 29 * 29 : (l == 7) ? 57 * 57
                               : (l == 0) ? 32 * 32 : (l == 1) ? 16 * 16 : 8 * 8) *
                  (size_t)nKL[l] * 32 * 2 * sizeof(_Float16);
    if (need > abytes) abytes = need;
  }
  _Float16* Abuf = (_Float16*)carve(abytes);  // shared im2col scratch (~240 MB)

  // deterministic state: re-zero hidden every call
  {
    int n = Bn * 128 * 8 * 8;
    zero_kernel<<<(n + 255) / 256, 256, 0, stream>>>(Hid, n);
  }
  // pack all weights once per launch (timestep-invariant)
  for (int l = 0; l < 8; ++l) {
    dim3 g(nKL[l], nTL[l]);
    pack_w_kernel<<<g, 32, 0, stream>>>(W[l], PK[l], CoutL[l], CinL[l], KkL[l], modeL[l]);
  }

  const int FRAME_N = Bn * 4096;
  float* outf = (float*)d_out;

  // layer = pack_a (im2col -> A fragments) + gemm_wmma (+fused epilogue)
#define RUN_LAYER(L, CinV, IHV, OHV, KkV, sdV, pqV, modeV, CoutV, inP, outP, addP)    \
  {                                                                                   \
    constexpr int nKc = ((CinV) * (KkV) * (KkV) + 31) >> 5;                           \
    const int M = Bn * (OHV) * (OHV);                                                 \
    pack_a_kernel<CinV, IHV, OHV, KkV, sdV, pqV, modeV>                               \
        <<<dim3(nKc, M / 16), 32, 0, stream>>>(inP, Abuf);                            \
    gemm_wmma_kernel<CoutV, OHV>                                                      \
        <<<dim3(M / 32, ((CoutV) + 15) / 16), 32, 0, stream>>>(                       \
            Abuf, PK[L], Bi[L], Gg[L], Gb[L], Gm[L], Gv[L], addP, outP, nKc);         \
  }

  for (int i = 0; i < NIN + NOUT - 1; ++i) {
    const float* frame;
    if (i < NIN) {
      gather_frame_kernel<<<(FRAME_N + 255) / 256, 256, 0, stream>>>(
          frames, Fbuf, i, T, FRAME_N);
      frame = Fbuf;
    } else {
      frame = outf + (size_t)(i - NIN) * FRAME_N;  // previous prediction
    }

    RUN_LAYER(0, 1,   64, 32, 9, 2, 4, 0, 32,  frame, E1, nullptr)       // en1
    RUN_LAYER(1, 32,  32, 16, 9, 2, 4, 0, 64,  E1,    E2, nullptr)       // en2
    RUN_LAYER(2, 64,  16, 8,  3, 2, 1, 0, 128, E2,    E3, nullptr)       // en3
    RUN_LAYER(3, 128, 8,  8,  3, 1, 1, 0, 128, E3,    H1, nullptr)       // hc1
    RUN_LAYER(4, 128, 8,  8,  3, 1, 1, 0, 128, H1,    Hid, Hid)          // hc2 (+state)
    RUN_LAYER(5, 128, 8,  15, 3, 2, 1, 1, 64,  Hid,   D1, nullptr)       // de1
    RUN_LAYER(6, 64,  15, 29, 3, 2, 1, 1, 32,  D1,    D2, nullptr)       // de2
    RUN_LAYER(7, 32,  29, 57, 3, 2, 1, 1, 81,  D2,    D3, nullptr)       // de3

    if (i >= NIN - 1) {
      float* pred = outf + (size_t)(i - (NIN - 1)) * FRAME_N;
      dfn_filter_kernel<<<(FRAME_N + 255) / 256, 256, 0, stream>>>(
          D3, frame, pred, Bn);
    }
  }
#undef RUN_LAYER
}
